// GATPolicyNetwork_17214228923073
// MI455X (gfx1250) — compile-verified, matrix operand order checked by reading, not run on hardware
//
#include <hip/hip_runtime.h>
#include <hip/hip_bf16.h>
#include <cstdint>
#include <cstddef>

typedef __attribute__((ext_vector_type(16))) _Float16 v16h;
typedef __attribute__((ext_vector_type(8)))  _Float16 v8h;
typedef __attribute__((ext_vector_type(8)))  float    v8f;
typedef __attribute__((ext_vector_type(4)))  float    v4f;

#define NNODES 50000
#define NEDGES 600000
#define ETOT   (NEDGES + NNODES)
#define DD     128
#define MT     64          // output rows per block in the WMMA GEMM
#define NGRAPH 64
#define NACT   64
#define NEG_SLOPE 0.2f

// ---------------------------------------------------------------------------
// helpers
// ---------------------------------------------------------------------------
__device__ __forceinline__ void edge_sd(const int* __restrict__ ei, int e,
                                        int& s, int& d) {
  if (e < NEDGES) { s = ei[e]; d = ei[NEDGES + e]; }
  else            { s = e - NEDGES; d = s; }   // self loops appended
}

// float atomic max via int/uint punning (monotone bit trick)
__device__ __forceinline__ void atomicMaxF(float* addr, float v) {
  if (v >= 0.0f) atomicMax((int*)addr, __float_as_int(v));
  else           atomicMin((unsigned int*)addr, __float_as_uint(v));
}

// ---------------------------------------------------------------------------
// C[M,128] = A[M,128] x W[128,128] via v_wmma_f32_16x16x32_f16
// block = 256 thr = 8 wave32. Block computes a 64x128 output tile:
//   - W staged transposed in LDS as f16 (32 KB): Bs[n*128+k] = W[k][n]
//   - A rows staged in LDS as f16 (16 KB):       As[m*128+k]
//   - wave w owns 16-col strip n0=16w; loops 4 M-tiles reusing its B frags
// 16 WMMAs per wave; f32->f16 conversion done once per element at staging.
// ---------------------------------------------------------------------------
__global__ __launch_bounds__(256)
void gemm128_wmma(const float* __restrict__ A, const float* __restrict__ W,
                  float* __restrict__ C, int M) {
  __shared__ __align__(16) _Float16 Bs[DD * DD];   // 32 KB
  __shared__ __align__(16) _Float16 As[MT * DD];   // 16 KB
  int tid  = threadIdx.x;
  int row0 = blockIdx.x * MT;

  // stage W transposed (f32 -> f16)
  for (int i = tid; i < DD * DD; i += 256) {
    int n = i >> 7, k = i & 127;
    Bs[i] = (_Float16)W[k * DD + n];
  }
  // stage A rows (f32 -> f16), edge rows clamped (stores are masked later)
  for (int i = tid * 4; i < MT * DD; i += 1024) {
    int m = i >> 7, k = i & 127;
    int gm = row0 + m;
    if (gm >= M) gm = M - 1;
    v4f f = *(const v4f*)(A + (size_t)gm * DD + k);
    As[i + 0] = (_Float16)f[0];
    As[i + 1] = (_Float16)f[1];
    As[i + 2] = (_Float16)f[2];
    As[i + 3] = (_Float16)f[3];
  }
  // warm L2 with the next block's A tile (global_prefetch_b8)
  if (row0 + MT < M)
    __builtin_prefetch(A + (size_t)(row0 + MT) * DD + tid * 32, 0, 1);
  __syncthreads();

  int wave  = tid >> 5;
  int lane  = tid & 31;
  int r     = lane & 15;                // row (A) / col (B) within tile
  int khalf = (lane >> 4) << 3;         // lanes 16-31 hold the K+8 half
  int n0    = wave << 4;
  const _Float16* Brow = Bs + (size_t)(n0 + r) * DD;

  v8f acc[4] = {{}, {}, {}, {}};
  #pragma unroll
  for (int k0 = 0; k0 < DD; k0 += 32) {
    // B fragment: elements 0-7 -> K = k0+khalf+0..7, 8-15 -> +16..23
    v8h b0 = *(const v8h*)(Brow + k0 + khalf);
    v8h b1 = *(const v8h*)(Brow + k0 + khalf + 16);
    v16h b;
    #pragma unroll
    for (int e = 0; e < 8; ++e) { b[e] = b0[e]; b[e + 8] = b1[e]; }
    #pragma unroll
    for (int mt = 0; mt < 4; ++mt) {
      const _Float16* Arow = As + (size_t)(mt * 16 + r) * DD;
      v8h a0 = *(const v8h*)(Arow + k0 + khalf);
      v8h a1 = *(const v8h*)(Arow + k0 + khalf + 16);
      v16h a;
      #pragma unroll
      for (int e = 0; e < 8; ++e) { a[e] = a0[e]; a[e + 8] = a1[e]; }
      acc[mt] = __builtin_amdgcn_wmma_f32_16x16x32_f16(
          false, a, false, b, (short)0, acc[mt], false, false);
    }
  }

  // D layout: VGPR i, lanes 0-15: M=i, N=lane; lanes 16-31: M=i+8, N=lane-16
  int n   = n0 + (lane & 15);
  int mbl = (lane >> 4) << 3;
  #pragma unroll
  for (int mt = 0; mt < 4; ++mt) {
    #pragma unroll
    for (int i = 0; i < 8; ++i) {
      int m = row0 + mt * 16 + mbl + i;
      if (m < M) C[(size_t)m * DD + n] = acc[mt][i];
    }
  }
}

// ---------------------------------------------------------------------------
// per-node attention projections: al_s[i]=h_i . a_src, al_d[i]=h_i . a_dst
// ---------------------------------------------------------------------------
__global__ __launch_bounds__(256)
void alproj_k(const float* __restrict__ h, const float* __restrict__ asrc,
              const float* __restrict__ adst, float* __restrict__ als,
              float* __restrict__ ald) {
  int i = blockIdx.x * blockDim.x + threadIdx.x;
  if (i >= NNODES) return;
  const float* row = h + (size_t)i * DD;
  float s = 0.0f, d = 0.0f;
  #pragma unroll 4
  for (int k = 0; k < DD; ++k) {
    float v = row[k];
    s += v * asrc[k];
    d += v * adst[k];
  }
  als[i] = s;
  ald[i] = d;
}

__global__ __launch_bounds__(256)
void fill_k(float* __restrict__ p, float v, int n) {
  int i = blockIdx.x * blockDim.x + threadIdx.x;
  if (i < n) p[i] = v;
}

__global__ __launch_bounds__(256)
void edge_max_k(const int* __restrict__ ei, const float* __restrict__ als,
                const float* __restrict__ ald, float* __restrict__ m) {
  int e = blockIdx.x * blockDim.x + threadIdx.x;
  if (e >= ETOT) return;
  int s, d; edge_sd(ei, e, s, d);
  float ev = als[s] + ald[d];
  ev = (ev >= 0.0f) ? ev : NEG_SLOPE * ev;
  atomicMaxF(&m[d], ev);
}

__global__ __launch_bounds__(256)
void edge_expsum_k(const int* __restrict__ ei, const float* __restrict__ als,
                   const float* __restrict__ ald, const float* __restrict__ m,
                   float* __restrict__ w, float* __restrict__ z) {
  int e = blockIdx.x * blockDim.x + threadIdx.x;
  if (e >= ETOT) return;
  int s, d; edge_sd(ei, e, s, d);
  float ev = als[s] + ald[d];
  ev = (ev >= 0.0f) ? ev : NEG_SLOPE * ev;
  float we = __expf(ev - m[d]);
  w[e] = we;
  atomicAdd(&z[d], we);
}

// one block per edge, 128 threads = feature dims; h[src] is L2-resident
__global__ __launch_bounds__(DD)
void edge_agg_k(const int* __restrict__ ei, const float* __restrict__ w,
                const float* __restrict__ z, const float* __restrict__ h,
                float* __restrict__ out) {
  int e = blockIdx.x;
  int s, d; edge_sd(ei, e, s, d);
  float alpha = w[e] / z[d];
  int t = threadIdx.x;
  atomicAdd(&out[(size_t)d * DD + t], alpha * h[(size_t)s * DD + t]);
}

__global__ __launch_bounds__(256)
void bias_act_k(float* __restrict__ h, const float* __restrict__ b, int relu) {
  size_t i = (size_t)blockIdx.x * blockDim.x + threadIdx.x;
  if (i >= (size_t)NNODES * DD) return;
  float v = h[i] + b[i & (DD - 1)];
  if (relu) v = v > 0.0f ? v : 0.0f;
  h[i] = v;
}

__global__ __launch_bounds__(DD)
void pool_k(const float* __restrict__ h, const int* __restrict__ batch,
            float* __restrict__ sums, float* __restrict__ cnts) {
  int i = blockIdx.x;
  int d = threadIdx.x;
  int g = batch[i];
  atomicAdd(&sums[(size_t)g * DD + d], h[(size_t)i * DD + d]);
  if (d == 0) atomicAdd(&cnts[g], 1.0f);
}

__global__ __launch_bounds__(NACT)
void head_k(const float* __restrict__ sums, const float* __restrict__ cnts,
            const float* __restrict__ Wh, const float* __restrict__ bh,
            float* __restrict__ out) {
  int g = blockIdx.x;
  int a = threadIdx.x;
  float c = cnts[g];
  c = c > 1.0f ? c : 1.0f;
  float acc = bh[a];
  #pragma unroll 4
  for (int k = 0; k < DD; ++k)
    acc += (sums[(size_t)g * DD + k] / c) * Wh[k * NACT + a];
  out[g * NACT + a] = acc;
}

// ---------------------------------------------------------------------------
extern "C" void kernel_launch(void* const* d_in, const int* in_sizes, int n_in,
                              void* d_out, int out_size, void* d_ws, size_t ws_size,
                              hipStream_t stream) {
  const float* x     = (const float*)d_in[0];
  const int*   ei    = (const int*)d_in[1];
  const int*   batch = (const int*)d_in[2];
  const float* Wl[3]  = {(const float*)d_in[3], (const float*)d_in[7],  (const float*)d_in[11]};
  const float* asl[3] = {(const float*)d_in[4], (const float*)d_in[8],  (const float*)d_in[12]};
  const float* adl[3] = {(const float*)d_in[5], (const float*)d_in[9],  (const float*)d_in[13]};
  const float* bl[3]  = {(const float*)d_in[6], (const float*)d_in[10], (const float*)d_in[14]};
  const float* Wh = (const float*)d_in[15];
  const float* bh = (const float*)d_in[16];
  float* out = (float*)d_out;

  float* ws   = (float*)d_ws;
  float* buf0 = ws;                                 // agg result / next input
  float* bufH = buf0 + (size_t)NNODES * DD;         // h = X @ W
  float* als  = bufH + (size_t)NNODES * DD;
  float* ald  = als + NNODES;
  float* mbuf = ald + NNODES;
  float* zbuf = mbuf + NNODES;
  float* wbuf = zbuf + NNODES;
  float* sums = wbuf + ETOT;
  float* cnts = sums + (size_t)NGRAPH * DD;

  const int gN    = (NNODES + 255) / 256;
  const int gND   = ((NNODES * DD) + 255) / 256;
  const int gE    = (ETOT + 255) / 256;
  const int gGemm = (NNODES + MT - 1) / MT;
  const float NEG_INF = -__builtin_huge_valf();

  const float* X = x;
  for (int l = 0; l < 3; ++l) {
    gemm128_wmma<<<gGemm, 256, 0, stream>>>(X, Wl[l], bufH, NNODES);
    alproj_k<<<gN, 256, 0, stream>>>(bufH, asl[l], adl[l], als, ald);
    fill_k<<<gND, 256, 0, stream>>>(buf0, 0.0f, NNODES * DD);
    fill_k<<<gN, 256, 0, stream>>>(mbuf, NEG_INF, NNODES);
    fill_k<<<gN, 256, 0, stream>>>(zbuf, 0.0f, NNODES);
    edge_max_k<<<gE, 256, 0, stream>>>(ei, als, ald, mbuf);
    edge_expsum_k<<<gE, 256, 0, stream>>>(ei, als, ald, mbuf, wbuf, zbuf);
    edge_agg_k<<<ETOT, DD, 0, stream>>>(ei, wbuf, zbuf, bufH, buf0);
    bias_act_k<<<gND, 256, 0, stream>>>(buf0, bl[l], l < 2 ? 1 : 0);
    X = buf0;
  }

  fill_k<<<(NGRAPH * DD + 255) / 256, 256, 0, stream>>>(sums, 0.0f, NGRAPH * DD);
  fill_k<<<1, 64, 0, stream>>>(cnts, 0.0f, NGRAPH);
  pool_k<<<NNODES, DD, 0, stream>>>(buf0, batch, sums, cnts);
  head_k<<<NGRAPH, NACT, 0, stream>>>(sums, cnts, Wh, bh, out);
}
